// Head_38749194944757
// MI455X (gfx1250) — compile-verified, matrix operand order checked by reading, not run on hardware
//
#include <hip/hip_runtime.h>
#include <hip/hip_bf16.h>

// ---------------- CDNA5 WMMA types ----------------
typedef __attribute__((ext_vector_type(16))) _Float16 v16h;
typedef __attribute__((ext_vector_type(8)))  _Float16 v8h;
typedef __attribute__((ext_vector_type(8)))  float    v8f;
typedef __attribute__((ext_vector_type(4)))  float    v4f;

#define D_MODEL 1024
#define D_HEAD  64
#define BATCH   4
#define TLEN    4096

static __device__ __forceinline__ v8f vzero8() {
    v8f z;
#pragma unroll
    for (int i = 0; i < 8; ++i) z[i] = 0.0f;
    return z;
}

// WMMA wrapper: D = A(16x32 f16) * B(32x16 f16) + C(16x16 f32)
static __device__ __forceinline__ v8f wmma_f16(v16h a, v16h b, v8f c) {
    return __builtin_amdgcn_wmma_f32_16x16x32_f16(
        /*neg_a=*/false, a, /*neg_b=*/false, b,
        /*c_mod=*/(short)0, c, /*reuse_a=*/false, /*reuse_b=*/false);
}

// Load a 16-element f16 fragment whose element axis is contiguous in memory:
// elems 0..7 at p, elems 8..15 at p+16 (the CDNA5 K(i)=i+8*((i>=8)+hi) split).
static __device__ __forceinline__ v16h load_frag(const _Float16* __restrict__ p) {
    v8h a0 = *(const v8h*)(p);
    v8h a1 = *(const v8h*)(p + 16);
    v16h f;
#pragma unroll
    for (int j = 0; j < 8; ++j) { f[j] = a0[j]; f[8 + j] = a1[j]; }
    return f;
}

// ============================================================
// Kernel 0: transpose weights f32 W[1024,64] -> f16 WT[64,1024].
// Makes the B-fragment element axis (K-dim) contiguous so the GEMM
// loads it with 16-byte v8h loads instead of 64 scalar b32 loads.
// Writes are fully coalesced; reads hit L2 (W is 256 KB).
// ============================================================
__global__ __launch_bounds__(256) void wtrans_kernel(
    const float* __restrict__ Wq, const float* __restrict__ Wk,
    const float* __restrict__ Wv, _Float16* __restrict__ wT)
{
    const float* W = (blockIdx.y == 0) ? Wq : (blockIdx.y == 1) ? Wk : Wv;
    _Float16*    o = wT + (size_t)blockIdx.y * D_MODEL * D_HEAD;
    const int idx = blockIdx.x * 256 + threadIdx.x;   // = n*1024 + k
    const int n = idx >> 10;
    const int kk = idx & 1023;
    o[idx] = (_Float16)W[(size_t)kk * D_HEAD + n];
}

// ============================================================
// Kernel 1: QKV projection.  X[16384,1024] (f32) x WT[64,1024] (f16)
// One wave computes a 16(M) x 64(N) tile; 8 waves per block.
// grid = (128, 3): y selects {Q,K,V}.
// Q,K stored row-major [tok][64]; V stored TRANSPOSED per batch
// [b][64][T] so the attention V-fragment loads contiguously too.
// ============================================================
__global__ __launch_bounds__(256) void qkv_proj_kernel(
    const float* __restrict__ x,
    const _Float16* __restrict__ wT,
    _Float16* __restrict__ qws, _Float16* __restrict__ kws,
    _Float16* __restrict__ vTws)
{
    const int lane = threadIdx.x & 31;
    const int wv   = threadIdx.x >> 5;
    const int lo   = lane & 15;
    const int hi   = lane >> 4;

    const _Float16* W = wT + (size_t)blockIdx.y * D_MODEL * D_HEAD; // [64][1024]

    const int mt  = blockIdx.x * 8 + wv;        // 16-row tile index
    const int row = mt * 16 + lo;               // this lane's A row
    const float* xrow = x + (size_t)row * D_MODEL;

    v8f acc[4];
#pragma unroll
    for (int t = 0; t < 4; ++t) acc[t] = vzero8();

    for (int kb = 0; kb < D_MODEL; kb += 32) {
        // ---- A fragment from x (f32 -> f16), contiguous 32B loads ----
        const float* xp = xrow + kb + 8 * hi;
        v4f x0 = *(const v4f*)(xp + 0);
        v4f x1 = *(const v4f*)(xp + 4);
        v4f x2 = *(const v4f*)(xp + 16);
        v4f x3 = *(const v4f*)(xp + 20);
        v16h a;
#pragma unroll
        for (int j = 0; j < 4; ++j) {
            a[j]      = (_Float16)x0[j];
            a[4 + j]  = (_Float16)x1[j];
            a[8 + j]  = (_Float16)x2[j];
            a[12 + j] = (_Float16)x3[j];
        }
        // ---- 4 N-tiles; B-fragment: lane col n, elems contiguous in k ----
#pragma unroll
        for (int t = 0; t < 4; ++t) {
            const int n = t * 16 + lo;
            v16h b = load_frag(W + (size_t)n * D_MODEL + kb + 8 * hi);
            acc[t] = wmma_f16(a, b, acc[t]);
        }
    }

    // C layout: VGPR r holds rows r (lanes 0-15) / r+8 (lanes 16-31), col = l&15
    if (blockIdx.y == 2) {
        // V: store transposed [b][h=64][T]
        const int b    = (mt * 16) / TLEN;
        const int tloc = mt * 16 - b * TLEN;
        _Float16* vT = vTws + (size_t)b * D_HEAD * TLEN;
#pragma unroll
        for (int t = 0; t < 4; ++t)
#pragma unroll
            for (int r = 0; r < 8; ++r)
                vT[(size_t)(t * 16 + lo) * TLEN + tloc + r + 8 * hi] =
                    (_Float16)acc[t][r];
    } else {
        _Float16* o = (blockIdx.y == 0) ? qws : kws;
#pragma unroll
        for (int t = 0; t < 4; ++t)
#pragma unroll
            for (int r = 0; r < 8; ++r)
                o[(size_t)(mt * 16 + r + 8 * hi) * D_HEAD + t * 16 + lo] =
                    (_Float16)acc[t][r];
    }
}

// ============================================================
// Kernel 2: causal flash attention on f16 Q/K/V^T, f32 accumulate.
// Swapped-operand form: S^T = K_tile(16x64) x Q^T(64x16).
//  - softmax over keys = per-lane reduce + shfl_xor(16)
//  - S^T C-layout IS the A-fragment layout for P in P@V (lane-local).
// One wave = 16 query rows; block = 8 waves = 128 rows; grid = (32, B).
// ============================================================
__global__ __launch_bounds__(256) void attn_kernel(
    const _Float16* __restrict__ qws, const _Float16* __restrict__ kws,
    const _Float16* __restrict__ vTws, float* __restrict__ out)
{
    const int lane = threadIdx.x & 31;
    const int w    = threadIdx.x >> 5;
    const int lo   = lane & 15;
    const int hi   = lane >> 4;

    const int b     = blockIdx.y;
    const int qbase = blockIdx.x * 128 + w * 16;   // first query row of this wave

    const _Float16* q  = qws  + (size_t)b * TLEN * D_HEAD;
    const _Float16* k  = kws  + (size_t)b * TLEN * D_HEAD;
    const _Float16* vT = vTws + (size_t)b * D_HEAD * TLEN;   // [64][T]

    // ---- Q^T B-fragments (fixed for whole key loop): lane col = query lo ----
    v16h bq[2];
#pragma unroll
    for (int hs = 0; hs < 2; ++hs)
        bq[hs] = load_frag(q + (size_t)(qbase + lo) * D_HEAD + hs * 32 + 8 * hi);

    float m_i = -1.0e30f;   // running row max (per lane's query = qbase+lo)
    float l_i = 0.0f;       // running row sum
    v8f O[4];
#pragma unroll
    for (int t = 0; t < 4; ++t) O[t] = vzero8();

    const int qg  = qbase + lo;             // this lane's global query row
    const int nkb = (qbase + 15) / 32 + 1;  // causal key-block bound

    for (int kb = 0; kb < nkb; ++kb) {
        const int kbase = kb * 32;

        // ---- S^T for two 16-key subtiles ----
        v8f C[2];
#pragma unroll
        for (int s = 0; s < 2; ++s) {
            C[s] = vzero8();
#pragma unroll
            for (int hs = 0; hs < 2; ++hs) {
                v16h ak = load_frag(
                    k + (size_t)(kbase + s * 16 + lo) * D_HEAD + hs * 32 + 8 * hi);
                C[s] = wmma_f16(ak, bq[hs], C[s]);
            }
        }

        // prefetch next key block's K rows / V^T columns (global_prefetch_b8)
        if (kb + 1 < nkb) {
            __builtin_prefetch(k + (size_t)(kbase + 32 + lo) * D_HEAD, 0, 1);
            __builtin_prefetch(vT + (size_t)(lane * 2) * TLEN + kbase + 32, 0, 1);
        }

        // ---- causal mask (only diagonal-straddling blocks) ----
        if (kbase + 31 > qbase) {
#pragma unroll
            for (int s = 0; s < 2; ++s)
#pragma unroll
                for (int r = 0; r < 8; ++r) {
                    const int kg = kbase + s * 16 + r + 8 * hi;
                    if (kg > qg) C[s][r] = -1.0e30f;
                }
        }

        // ---- online softmax (rows = queries; lane l owns query l&15) ----
        float mloc = m_i;
#pragma unroll
        for (int r = 0; r < 8; ++r)
            mloc = fmaxf(mloc, fmaxf(C[0][r], C[1][r]));
        mloc = fmaxf(mloc, __shfl_xor(mloc, 16));
        const float mnew  = mloc;
        const float alpha = __expf(m_i - mnew);

        float p0[8], p1[8], rsum = 0.0f;
#pragma unroll
        for (int r = 0; r < 8; ++r) {
            p0[r] = __expf(C[0][r] - mnew);
            p1[r] = __expf(C[1][r] - mnew);
            rsum += p0[r] + p1[r];
        }
        rsum += __shfl_xor(rsum, 16);
        l_i = l_i * alpha + rsum;
        m_i = mnew;

        // ---- P A-fragment: pure lane-local repack of C0/C1 ----
        v16h pa;
#pragma unroll
        for (int r = 0; r < 8; ++r) {
            pa[r]     = (_Float16)p0[r];
            pa[8 + r] = (_Float16)p1[r];
        }

        // ---- rescale O rows by alpha (row m's alpha lives in lane m) ----
#pragma unroll
        for (int r = 0; r < 8; ++r) {
            const float ar = __shfl(alpha, r + 8 * hi);
#pragma unroll
            for (int t = 0; t < 4; ++t) O[t][r] *= ar;
        }

        // ---- O += P(16x32) @ V(32x64); V^T makes elems contiguous ----
#pragma unroll
        for (int t = 0; t < 4; ++t) {
            v16h bv = load_frag(
                vT + (size_t)(t * 16 + lo) * TLEN + kbase + 8 * hi);
            O[t] = wmma_f16(pa, bv, O[t]);
        }
    }

    // ---- epilogue: divide by l_i and the post-softmax 1/sqrt(64)=1/8 ----
    const float inv = 1.0f / (l_i * 8.0f);
    float* ob = out + ((size_t)b * TLEN + qbase) * D_HEAD;
#pragma unroll
    for (int r = 0; r < 8; ++r) {
        const float sr = __shfl(inv, r + 8 * hi);
        const int   m  = r + 8 * hi;
#pragma unroll
        for (int t = 0; t < 4; ++t)
            ob[(size_t)m * D_HEAD + t * 16 + lo] = O[t][r] * sr;
    }
}

// ============================================================
extern "C" void kernel_launch(void* const* d_in, const int* in_sizes, int n_in,
                              void* d_out, int out_size, void* d_ws, size_t ws_size,
                              hipStream_t stream) {
    const float* x  = (const float*)d_in[0];
    const float* Wq = (const float*)d_in[1];
    const float* Wk = (const float*)d_in[2];
    const float* Wv = (const float*)d_in[3];
    float* out = (float*)d_out;

    const size_t n = (size_t)BATCH * TLEN * D_HEAD;   // 1,048,576 halves each
    _Float16* qws  = (_Float16*)d_ws;
    _Float16* kws  = qws + n;
    _Float16* vTws = kws + n;
    _Float16* wT   = vTws + n;                        // 3 x [64][1024] f16

    // Weight transpose: 65536 elems per matrix, 256 blocks; y selects Q/K/V
    wtrans_kernel<<<dim3(D_MODEL * D_HEAD / 256, 3), 256, 0, stream>>>(
        Wq, Wk, Wv, wT);

    // QKV projection: 16384/16 = 1024 wave-tiles, 8 waves/block -> 128 blocks
    qkv_proj_kernel<<<dim3(128, 3), 256, 0, stream>>>(x, wT, qws, kws, vTws);

    // Flash attention: 4096/128 = 32 blocks per batch
    attn_kernel<<<dim3(TLEN / 128, BATCH), 256, 0, stream>>>(qws, kws, vTws, out);
}
